// GCNet_66984309948590
// MI455X (gfx1250) — matrix operand order repaired and checked by reading, not compiled, not run on hardware
//
#include <hip/hip_runtime.h>
#include <hip/hip_fp16.h>

typedef __attribute__((ext_vector_type(16))) _Float16 v16h;
typedef __attribute__((ext_vector_type(8)))  float    v8f;

static __device__ __forceinline__ float inv_deg(float d) {
    return d > 0.f ? 1.f / fmaxf(d, 1.f) : 0.f;
}
// post-ReLU values are >= +0, so uint bit ordering == float ordering
static __device__ __forceinline__ void atomicMaxPosF(float* p, float v) {
    atomicMax((unsigned int*)p, __float_as_uint(v));
}

// ---------------- utility fills (vectorized) ----------------
__global__ void k_zero_f(float* __restrict__ p, long long cnt) {
    long long i4 = ((long long)blockIdx.x * blockDim.x + threadIdx.x) * 4;
    if (i4 + 4 <= cnt) {
        *(float4*)(p + i4) = make_float4(0.f, 0.f, 0.f, 0.f);
    } else {
        for (long long j = i4; j < cnt; ++j) p[j] = 0.f;
    }
}
__global__ void k_fill_i(int* __restrict__ p, int v, long long cnt) {
    long long i4 = ((long long)blockIdx.x * blockDim.x + threadIdx.x) * 4;
    if (i4 + 4 <= cnt) {
        *(int4*)(p + i4) = make_int4(v, v, v, v);
    } else {
        for (long long j = i4; j < cnt; ++j) p[j] = v;
    }
}

// ---------------- GNN1 dense transform: one wave = 16 nodes ----------------
// A tile: 16x32 f16 (x rows, K padded 5->32). B tiles: 32x16 f16 (W1 / W1r padded).
// A layout: lanes 0-15 hold row m=lane, halves 0..7 -> k=0..7, 8..15 -> k=16..23;
//           lanes 16-31 hold row m=lane-16, k = 8..15 / 24..31.
// B layout: lane%16 = output column n, halves follow the A k order (rows k>=5 zero).
__global__ __launch_bounds__(256)
void k_gnn1_wmma(const float* __restrict__ x, const float* __restrict__ W,
                 const float* __restrict__ Wr, const float* __restrict__ b,
                 float* __restrict__ xw, float* __restrict__ xr, int n)
{
    const int lane = threadIdx.x & 31;
    const int wave = threadIdx.x >> 5;
    const int tile = blockIdx.x * 8 + wave;
    const long long row0 = (long long)tile * 16;
    if (row0 >= n) return;
    const bool full = (row0 + 16 <= n);

    const int m   = lane & 15;
    const int khi = (lane >> 4) ? 8 : 0;
    v16h a = {};
    if (full) {
        // only lanes 0-15 (khi==0) touch k<5; loads unconditional per active lane
        if (lane < 16) {
            const float* xrow = x + (row0 + m) * 5;
            #pragma unroll
            for (int h = 0; h < 5; ++h) a[h] = (_Float16)xrow[h];
        }
    } else {
        #pragma unroll
        for (int h = 0; h < 16; ++h) {
            int k = ((h < 8) ? h : (h - 8) + 16) + khi;
            long long r = row0 + m;
            float v = (k < 5 && r < n) ? x[r * 5 + k] : 0.f;
            a[h] = (_Float16)v;
        }
    }
    v16h bw, br;
    #pragma unroll
    for (int h = 0; h < 16; ++h) {
        float vw = 0.f, vr = 0.f;
        { int kk = ((h < 8) ? h : h + 8) + khi; if (kk < 5) { vw = W[kk * 16 + (lane & 15)]; vr = Wr[kk * 16 + (lane & 15)]; } }
        bw[h] = (_Float16)vw; br[h] = (_Float16)vr;
    }
    v8f c0 = {}; v8f c1 = {};
    c0 = __builtin_amdgcn_wmma_f32_16x16x32_f16(false, a, false, bw, (short)0, c0, false, false);
    c1 = __builtin_amdgcn_wmma_f32_16x16x32_f16(false, a, false, br, (short)0, c1, false, false);

    const int colN  = lane & 15;
    const int rbase = (lane >> 4) * 8;
    const float bias = b[colN];
    if (full) {
        float* pw = xw + (row0 + rbase) * 16 + colN;
        float* pr = xr + (row0 + rbase) * 16 + colN;
        #pragma unroll
        for (int v = 0; v < 8; ++v) {
            pw[v * 16] = c0[v];
            pr[v * 16] = c1[v] + bias;
        }
    } else {
        #pragma unroll
        for (int v = 0; v < 8; ++v) {
            long long r = row0 + rbase + v;
            if (r < n) {
                xw[r * 16 + colN] = c0[v];
                xr[r * 16 + colN] = c1[v] + bias;
            }
        }
    }
}

// ---------------- GNN2 dense transform: xp(16) -> 32 features ----------------
__global__ __launch_bounds__(256)
void k_gnn2_wmma(const float* __restrict__ xp, const float* __restrict__ W,
                 const float* __restrict__ Wr, const float* __restrict__ b,
                 float* __restrict__ xw, float* __restrict__ xr, int n)
{
    const int lane = threadIdx.x & 31;
    const int wave = threadIdx.x >> 5;
    const int tile = blockIdx.x * 8 + wave;
    const long long row0 = (long long)tile * 16;
    if (row0 >= n) return;
    const bool full = (row0 + 16 <= n);

    const int m   = lane & 15;
    const int khi = (lane >> 4) ? 8 : 0;
    v16h a = {};
    if (full) {
        // lanes 0-15 load k=0..7, lanes 16-31 load k=8..15: no divergence at all
        const float* xrow = xp + (row0 + m) * 16 + khi;
        #pragma unroll
        for (int h = 0; h < 8; ++h) a[h] = (_Float16)xrow[h];
    } else {
        #pragma unroll
        for (int h = 0; h < 16; ++h) {
            int k = ((h < 8) ? h : (h - 8) + 16) + khi;   // K padded 16 -> 32
            long long r = row0 + m;
            float v = (k < 16 && r < n) ? xp[r * 16 + k] : 0.f;
            a[h] = (_Float16)v;
        }
    }
    const int colN  = lane & 15;
    const int rbase = (lane >> 4) * 8;
    #pragma unroll
    for (int half = 0; half < 2; ++half) {
        const int off = half * 16;
        v16h bw, br;
        #pragma unroll
        for (int h = 0; h < 16; ++h) {
            float vw = 0.f, vr = 0.f;
            { int kk = ((h < 8) ? h : h + 8) + khi; if (kk < 16) { vw = W[kk * 32 + off + (lane & 15)]; vr = Wr[kk * 32 + off + (lane & 15)]; } }
            bw[h] = (_Float16)vw; br[h] = (_Float16)vr;
        }
        v8f c0 = {}; v8f c1 = {};
        c0 = __builtin_amdgcn_wmma_f32_16x16x32_f16(false, a, false, bw, (short)0, c0, false, false);
        c1 = __builtin_amdgcn_wmma_f32_16x16x32_f16(false, a, false, br, (short)0, c1, false, false);
        const float bias = b[off + colN];
        if (full) {
            float* pw = xw + (row0 + rbase) * 32 + off + colN;
            float* pr = xr + (row0 + rbase) * 32 + off + colN;
            #pragma unroll
            for (int v = 0; v < 8; ++v) {
                pw[v * 32] = c0[v];
                pr[v * 32] = c1[v] + bias;
            }
        } else {
            #pragma unroll
            for (int v = 0; v < 8; ++v) {
                long long r = row0 + rbase + v;
                if (r < n) {
                    xw[r * 32 + off + colN] = c0[v];
                    xr[r * 32 + off + colN] = c1[v] + bias;
                }
            }
        }
    }
}

// ---------------- edge aggregation (mean messages at target=col) ----------------
__global__ void k_edge_agg1(const int* __restrict__ row, const int* __restrict__ col,
                            const float* __restrict__ xw, float* __restrict__ s,
                            float* __restrict__ cnt, int E)
{
    int e = blockIdx.x * blockDim.x + threadIdx.x;
    if (e >= E) return;
    long long r = row[e], c = col[e];
    const float* src = xw + r * 16;
    float* dst = s + c * 16;
    __builtin_prefetch(src, 0, 1);          // global_prefetch_b8: hide gather latency
    #pragma unroll
    for (int f = 0; f < 16; ++f) atomicAdd(&dst[f], src[f]);
    atomicAdd(&cnt[c], 1.f);
}

__global__ void k_x1(float* __restrict__ s, const float* __restrict__ xr,
                     const float* __restrict__ cnt, int n)
{
    int i = blockIdx.x * blockDim.x + threadIdx.x;
    if (i >= n) return;
    float inv = 1.f / fmaxf(cnt[i], 1.f);
    float4* sp = (float4*)(s + (long long)i * 16);
    const float4* rp = (const float4*)(xr + (long long)i * 16);
    #pragma unroll
    for (int q = 0; q < 4; ++q) {
        float4 sv = sp[q], rv = rp[q];
        sv.x = fmaxf(sv.x * inv + rv.x, 0.f);   // leaky(relu(z),0.1) == relu(z)
        sv.y = fmaxf(sv.y * inv + rv.y, 0.f);
        sv.z = fmaxf(sv.z * inv + rv.z, 0.f);
        sv.w = fmaxf(sv.w * inv + rv.w, 0.f);
        sp[q] = sv;
    }
}

// ---------------- graclus matching, level 1 (attr = |x[r,0]-x[c,0]|) ----------------
__global__ void k_deg1(const int* __restrict__ row, float* __restrict__ deg, int E) {
    int e = blockIdx.x * blockDim.x + threadIdx.x;
    if (e < E) atomicAdd(&deg[row[e]], 1.f);
}
__global__ void k_best1(const float* __restrict__ x, const int* __restrict__ row,
                        const int* __restrict__ col, const float* __restrict__ deg,
                        float* __restrict__ best, int E)
{
    int e = blockIdx.x * blockDim.x + threadIdx.x;
    if (e >= E) return;
    int r = row[e], c = col[e];
    float w = fabsf(x[(long long)r * 5] - x[(long long)c * 5]) * (inv_deg(deg[r]) + inv_deg(deg[c]));
    atomicMaxPosF(&best[r], w);
}
__global__ void k_partner1(const float* __restrict__ x, const int* __restrict__ row,
                           const int* __restrict__ col, const float* __restrict__ deg,
                           const float* __restrict__ best, int* __restrict__ partner, int E)
{
    int e = blockIdx.x * blockDim.x + threadIdx.x;
    if (e >= E) return;
    int r = row[e], c = col[e];
    float w = fabsf(x[(long long)r * 5] - x[(long long)c * 5]) * (inv_deg(deg[r]) + inv_deg(deg[c]));
    if (w == best[r]) atomicMin(&partner[r], c);
}
__global__ void k_cluster(const int* __restrict__ partner, int* __restrict__ cl, int n) {
    int i = blockIdx.x * blockDim.x + threadIdx.x;
    if (i >= n) return;
    int p = partner[i]; if (p >= n || p < 0) p = i;
    int pp = partner[p]; if (pp >= n || pp < 0) pp = p;
    cl[i] = (pp == i) ? min(i, p) : i;
}

// ---------------- pooling level 1: cluster max of x1, mean of pos ----------------
__global__ void k_pool1(const float* __restrict__ x1, const float* __restrict__ pos,
                        const int* __restrict__ c1, float* __restrict__ xp,
                        float* __restrict__ cnt1, float* __restrict__ posp, int n)
{
    int i = blockIdx.x * blockDim.x + threadIdx.x;
    if (i >= n) return;
    long long c = c1[i];
    const float* src = x1 + (long long)i * 16;
    float* dst = xp + c * 16;
    __builtin_prefetch(src, 0, 1);
    #pragma unroll
    for (int f = 0; f < 16; ++f) atomicMaxPosF(&dst[f], src[f]);
    atomicAdd(&cnt1[c], 1.f);
    atomicAdd(&posp[c * 2 + 0], pos[(long long)i * 2 + 0]);
    atomicAdd(&posp[c * 2 + 1], pos[(long long)i * 2 + 1]);
}
__global__ void k_pool1_fin(const int* __restrict__ c1, const float* __restrict__ cnt1,
                            float* __restrict__ xp, float* __restrict__ posp, int n)
{
    int i = blockIdx.x * blockDim.x + threadIdx.x;
    if (i >= n) return;
    float cn = fmaxf(cnt1[i], 1.f);
    posp[(long long)i * 2 + 0] /= cn;
    posp[(long long)i * 2 + 1] /= cn;
    if (c1[i] != i) {
        float4* p = (float4*)(xp + (long long)i * 16);
        #pragma unroll
        for (int q = 0; q < 4; ++q) p[q] = make_float4(0.f, 0.f, 0.f, 0.f);
    }
}

// ---------------- GNN2 edge aggregation on coarsened edges ----------------
__global__ void k_edge_agg2(const int* __restrict__ row, const int* __restrict__ col,
                            const int* __restrict__ c1, const float* __restrict__ xw,
                            float* __restrict__ s, float* __restrict__ cnt, int E)
{
    int e = blockIdx.x * blockDim.x + threadIdx.x;
    if (e >= E) return;
    long long r2 = c1[row[e]], c2 = c1[col[e]];
    if (r2 == c2) return;
    const float* src = xw + r2 * 32;
    float* dst = s + c2 * 32;
    __builtin_prefetch(src, 0, 1);
    #pragma unroll
    for (int f = 0; f < 32; ++f) atomicAdd(&dst[f], src[f]);
    atomicAdd(&cnt[c2], 1.f);
}
__global__ void k_x2(float* __restrict__ s, const float* __restrict__ xr,
                     const float* __restrict__ cnt, int n)
{
    int i = blockIdx.x * blockDim.x + threadIdx.x;
    if (i >= n) return;
    float inv = 1.f / fmaxf(cnt[i], 1.f);
    float4* sp = (float4*)(s + (long long)i * 32);
    const float4* rp = (const float4*)(xr + (long long)i * 32);
    #pragma unroll
    for (int q = 0; q < 8; ++q) {
        float4 sv = sp[q], rv = rp[q];
        sv.x = fmaxf(sv.x * inv + rv.x, 0.f);
        sv.y = fmaxf(sv.y * inv + rv.y, 0.f);
        sv.z = fmaxf(sv.z * inv + rv.z, 0.f);
        sv.w = fmaxf(sv.w * inv + rv.w, 0.f);
        sp[q] = sv;
    }
}

// ---------------- matching level 2 (attr = ||posp[r2]-posp[c2]||) ----------------
__global__ void k_deg2(const int* __restrict__ row, const int* __restrict__ col,
                       const int* __restrict__ c1, float* __restrict__ deg, int E)
{
    int e = blockIdx.x * blockDim.x + threadIdx.x;
    if (e >= E) return;
    int r2 = c1[row[e]], c2 = c1[col[e]];
    if (r2 != c2) atomicAdd(&deg[r2], 1.f);
}
static __device__ __forceinline__ float cut2(const float* posp, int r2, int c2,
                                             const float* deg) {
    float dx = posp[(long long)r2 * 2 + 0] - posp[(long long)c2 * 2 + 0];
    float dy = posp[(long long)r2 * 2 + 1] - posp[(long long)c2 * 2 + 1];
    float attr = sqrtf(dx * dx + dy * dy + 1e-12f);
    return attr * (inv_deg(deg[r2]) + inv_deg(deg[c2]));
}
__global__ void k_best2(const int* __restrict__ row, const int* __restrict__ col,
                        const int* __restrict__ c1, const float* __restrict__ posp,
                        const float* __restrict__ deg, float* __restrict__ best, int E)
{
    int e = blockIdx.x * blockDim.x + threadIdx.x;
    if (e >= E) return;
    int r2 = c1[row[e]], c2 = c1[col[e]];
    if (r2 == c2) return;
    atomicMaxPosF(&best[r2], cut2(posp, r2, c2, deg));
}
__global__ void k_partner2(const int* __restrict__ row, const int* __restrict__ col,
                           const int* __restrict__ c1, const float* __restrict__ posp,
                           const float* __restrict__ deg, const float* __restrict__ best,
                           int* __restrict__ partner, int E)
{
    int e = blockIdx.x * blockDim.x + threadIdx.x;
    if (e >= E) return;
    int r2 = c1[row[e]], c2 = c1[col[e]];
    if (r2 == c2) return;
    if (cut2(posp, r2, c2, deg) == best[r2]) atomicMin(&partner[r2], c2);
}

// ---------------- pooling level 2 + global mean ----------------
__global__ void k_pool2(const float* __restrict__ x2, const int* __restrict__ c1,
                        const int* __restrict__ c2, float* __restrict__ x3,
                        float* __restrict__ cnt3, int n)
{
    int i = blockIdx.x * blockDim.x + threadIdx.x;
    if (i >= n) return;
    if (c1[i] != i) return;                 // valid1 only
    long long j = c2[i];
    const float* src = x2 + (long long)i * 32;
    float* dst = x3 + j * 32;
    __builtin_prefetch(src, 0, 1);
    #pragma unroll
    for (int f = 0; f < 32; ++f) atomicMaxPosF(&dst[f], src[f]);
    atomicAdd(&cnt3[j], 1.f);
}
__global__ void k_reduce(const float* __restrict__ x3, const float* __restrict__ cnt3,
                         float* __restrict__ acc, int n)
{
    int i = blockIdx.x * blockDim.x + threadIdx.x;
    if (i >= n) return;
    if (cnt3[i] <= 0.f) return;             // valid2
    const float* src = x3 + (long long)i * 32;
    #pragma unroll
    for (int f = 0; f < 32; ++f) atomicAdd(&acc[f], src[f]);
    atomicAdd(&acc[32], 1.f);
}
__global__ void k_mlp(const float* __restrict__ acc, const float* __restrict__ l1w,
                      const float* __restrict__ l1b, const float* __restrict__ l2w,
                      const float* __restrict__ l2b, float* __restrict__ out)
{
    if (threadIdx.x != 0 || blockIdx.x != 0) return;
    float denom = fmaxf(acc[32], 1.f);
    float p[32];
    for (int f = 0; f < 32; ++f) p[f] = acc[f] / denom;
    float h[8];
    for (int o = 0; o < 8; ++o) {
        float z = l1b[o];
        for (int f = 0; f < 32; ++f) z += p[f] * l1w[f * 8 + o];
        h[o] = z > 0.f ? z : 0.1f * z;
    }
    for (int o = 0; o < 2; ++o) {
        float z = l2b[o];
        for (int k = 0; k < 8; ++k) z += h[k] * l2w[k * 2 + o];
        out[o] = z;
    }
}

// =====================================================================
extern "C" void kernel_launch(void* const* d_in, const int* in_sizes, int n_in,
                              void* d_out, int out_size, void* d_ws, size_t ws_size,
                              hipStream_t stream)
{
    (void)n_in; (void)out_size; (void)ws_size;
    const float* x    = (const float*)d_in[0];
    const float* pos  = (const float*)d_in[1];
    const int*   ei   = (const int*)  d_in[2];
    const float* W1   = (const float*)d_in[3];
    const float* W1r  = (const float*)d_in[4];
    const float* b1   = (const float*)d_in[5];
    const float* W2   = (const float*)d_in[6];
    const float* W2r  = (const float*)d_in[7];
    const float* b2   = (const float*)d_in[8];
    const float* l1w  = (const float*)d_in[9];
    const float* l1b  = (const float*)d_in[10];
    const float* l2w  = (const float*)d_in[11];
    const float* l2b  = (const float*)d_in[12];
    float* out = (float*)d_out;

    const int n = in_sizes[0] / 5;
    const int E = in_sizes[2] / 2;
    const int* row = ei;
    const int* col = ei + E;
    const long long N = n;

    // workspace carve-out (floats), heavy buffer reuse
    float* A   = (float*)d_ws;      // N*32 : xw1(stride16) -> xw2(stride32) -> x3(stride32)
    float* Bb  = A  + N * 32;       // N*32 : xr1 -> xr2
    float* C   = Bb + N * 32;       // N*32 : s1/x1(stride16) -> s2/x2(stride32)
    float* D   = C  + N * 32;       // N*16 : xp
    float* P   = D  + N * 16;       // N*2  : posp
    float* G0  = P  + N * 2;        // N : cntE1 / deg1 / cnt1 / deg2 / cnt3
    float* G1  = G0 + N;            // N : best1 / cntE2 / best2
    int*   G2  = (int*)(G1 + N);    // N : partner
    int*   G3  = G2 + N;            // N : c1
    int*   G4  = G3 + N;            // N : c2
    float* ACC = (float*)(G4 + N);  // 64 : global sums

    const int T = 256;
    dim3 gN((n + T - 1) / T), gE((E + T - 1) / T), bT(T);
    const int tiles = (n + 15) / 16;
    dim3 gW((tiles + 7) / 8);
    auto gz = [&](long long cnt) {           // vectorized fill: one thread per 4 elems
        long long th = (cnt + 3) / 4;
        return dim3((unsigned)((th + T - 1) / T));
    };

    // ---- GNN1 dense (WMMA) ----
    k_gnn1_wmma<<<gW, bT, 0, stream>>>(x, W1, W1r, b1, A, Bb, n);
    // ---- GNN1 mean aggregation ----
    k_zero_f<<<gz(N * 16), bT, 0, stream>>>(C, N * 16);
    k_zero_f<<<gz(N), bT, 0, stream>>>(G0, N);
    k_edge_agg1<<<gE, bT, 0, stream>>>(row, col, A, C, G0, E);
    k_x1<<<gN, bT, 0, stream>>>(C, Bb, G0, n);
    // ---- matching level 1 ----
    k_zero_f<<<gz(N), bT, 0, stream>>>(G0, N);
    k_deg1<<<gE, bT, 0, stream>>>(row, G0, E);
    k_zero_f<<<gz(N), bT, 0, stream>>>(G1, N);
    k_best1<<<gE, bT, 0, stream>>>(x, row, col, G0, G1, E);
    k_fill_i<<<gz(N), bT, 0, stream>>>(G2, n, N);
    k_partner1<<<gE, bT, 0, stream>>>(x, row, col, G0, G1, G2, E);
    k_cluster<<<gN, bT, 0, stream>>>(G2, G3, n);
    // ---- pool level 1 ----
    k_zero_f<<<gz(N * 16), bT, 0, stream>>>(D, N * 16);
    k_zero_f<<<gz(N), bT, 0, stream>>>(G0, N);
    k_zero_f<<<gz(N * 2), bT, 0, stream>>>(P, N * 2);
    k_pool1<<<gN, bT, 0, stream>>>(C, pos, G3, D, G0, P, n);
    k_pool1_fin<<<gN, bT, 0, stream>>>(G3, G0, D, P, n);
    // ---- GNN2 dense (WMMA) ----
    k_gnn2_wmma<<<gW, bT, 0, stream>>>(D, W2, W2r, b2, A, Bb, n);
    // ---- GNN2 mean aggregation on coarsened edges ----
    k_zero_f<<<gz(N * 32), bT, 0, stream>>>(C, N * 32);
    k_zero_f<<<gz(N), bT, 0, stream>>>(G1, N);
    k_edge_agg2<<<gE, bT, 0, stream>>>(row, col, G3, A, C, G1, E);
    k_x2<<<gN, bT, 0, stream>>>(C, Bb, G1, n);
    // ---- matching level 2 ----
    k_zero_f<<<gz(N), bT, 0, stream>>>(G0, N);
    k_deg2<<<gE, bT, 0, stream>>>(row, col, G3, G0, E);
    k_zero_f<<<gz(N), bT, 0, stream>>>(G1, N);
    k_best2<<<gE, bT, 0, stream>>>(row, col, G3, P, G0, G1, E);
    k_fill_i<<<gz(N), bT, 0, stream>>>(G2, n, N);
    k_partner2<<<gE, bT, 0, stream>>>(row, col, G3, P, G0, G1, G2, E);
    k_cluster<<<gN, bT, 0, stream>>>(G2, G4, n);
    // ---- pool level 2 + global mean + MLP head ----
    k_zero_f<<<gz(N * 32), bT, 0, stream>>>(A, N * 32);
    k_zero_f<<<gz(N), bT, 0, stream>>>(G0, N);
    k_pool2<<<gN, bT, 0, stream>>>(C, G3, G4, A, G0, n);
    k_zero_f<<<dim3(1), dim3(64), 0, stream>>>(ACC, 64);
    k_reduce<<<gN, bT, 0, stream>>>(A, G0, ACC, n);
    k_mlp<<<dim3(1), dim3(1), 0, stream>>>(ACC, l1w, l1b, l2w, l2b, out);
}
